// TrajectoryPredictor_79199196938369
// MI455X (gfx1250) — matrix operand order repaired and checked by reading, not compiled, not run on hardware
//
#include <hip/hip_runtime.h>
#include <hip/hip_bf16.h>

// ---------------------------------------------------------------------------
// TrajectoryPredictor on MI455X (gfx1250, wave32, WMMA)
//   LSTM encode  : WMMA f16->f32, fragment-major weights in LDS
//   GATv2 x2     : WMMA node transforms + fused softmax/scatter edge passes
//   Pool fused into layer-2 finalize; tiny decoder per graph
// ---------------------------------------------------------------------------

typedef __attribute__((ext_vector_type(16))) _Float16 v16h;
typedef __attribute__((ext_vector_type(8)))  _Float16 v8h;
typedef __attribute__((ext_vector_type(8)))  float    v8f;

#define H 64
#define G4 256          // 4*H
#define T_STEPS 20
#define HORIZON 60      // 30*2

__device__ __forceinline__ v8f wmma_f16(v16h a, v16h b, v8f c) {
    return __builtin_amdgcn_wmma_f32_16x16x32_f16(
        false, a, false, b, (short)0, c, false, false);
}

__device__ __forceinline__ float sigmoidf_(float v) {
    return 1.0f / (1.0f + __expf(-v));
}

// float -> order-preserving uint key (for atomicMax-based segment max)
__device__ __forceinline__ unsigned f2ord(float f) {
    unsigned u = __float_as_uint(f);
    return (u & 0x80000000u) ? ~u : (u | 0x80000000u);
}
__device__ __forceinline__ float ord2f(unsigned u) {
    return (u & 0x80000000u) ? __uint_as_float(u & 0x7fffffffu)
                             : __uint_as_float(~u);
}

#define CAT16(lo, hi) __builtin_shufflevector(lo, hi, 0,1,2,3,4,5,6,7,8,9,10,11,12,13,14,15)

// ---------------------------------------------------------------------------
// Kernel 1: LSTM encoder.  2 waves/block, 16 nodes per wave.
// gates = x_t@W_ih.T + h@W_hh.T + b ; recurrent matmul via WMMA.
// W_hh stored in LDS pre-swizzled to WMMA-B fragment layout:
//   sBfrag[ct][chunk][lane][j] = W_hh[g=ct*16+(lane&15)][k=chunk*32+((lane<16)?0:8)+(j&7)+(j>>3)*16]
// ---------------------------------------------------------------------------
#define LSTM_WAVES 2
__global__ __launch_bounds__(LSTM_WAVES * 32)
void lstm_kernel(const float* __restrict__ history,
                 const float* __restrict__ W_ih, const float* __restrict__ W_hh,
                 const float* __restrict__ b_ih, const float* __restrict__ b_hh,
                 float* __restrict__ x_out, int N)
{
    __shared__ _Float16 sBfrag[16][2][32][16];       // 32KB fragment-major W_hh^T
    __shared__ float    sWih[G4 * 2];                // W_ih row-major [g][d]
    __shared__ float    sB[G4];                      // b_ih + b_hh
    __shared__ _Float16 sH[LSTM_WAVES][16][H];       // h tile as f16 (WMMA A source)
    __shared__ _Float16 sG[LSTM_WAVES][16][G4];      // gates tile
    __shared__ float    sX[LSTM_WAVES][16][2];       // x_t tile

    const int tid = threadIdx.x;
    // cooperative weight staging (once per block), fragment-major
    for (int idx = tid; idx < 16 * 2 * 32 * 16; idx += LSTM_WAVES * 32) {
        int j     = idx & 15;
        int lane_ = (idx >> 4) & 31;
        int chunk = (idx >> 9) & 1;
        int ct    = idx >> 10;
        int col   = ct * 16 + (lane_ & 15);
        int k     = chunk * 32 + ((lane_ < 16) ? 0 : 8) + (j & 7) + (j >> 3) * 16;
        sBfrag[ct][chunk][lane_][j] = (_Float16)W_hh[col * H + k];
    }
    for (int idx = tid; idx < G4 * 2; idx += LSTM_WAVES * 32) sWih[idx] = W_ih[idx];
    for (int idx = tid; idx < G4; idx += LSTM_WAVES * 32) sB[idx] = b_ih[idx] + b_hh[idx];
    for (int idx = tid; idx < LSTM_WAVES * 16 * H; idx += LSTM_WAVES * 32)
        ((_Float16*)sH)[idx] = (_Float16)0.0f;
    __syncthreads();

    const int wave = tid >> 5, lane = tid & 31;
    const int base = (blockIdx.x * LSTM_WAVES + wave) * 16;
    const int nsel   = lane & 15;
    const int rowoff = (lane < 16) ? 0 : 8;          // C-frag row offset
    const int kb     = (lane < 16) ? 0 : 8;          // A-frag K base

    // per-lane cell state: node = nsel, cells cc0..cc0+31 (fully unrolled)
    const int cc0 = (lane >> 4) * 32;
    float creg[32];
#pragma unroll
    for (int j = 0; j < 32; j++) creg[j] = 0.0f;

    for (int t = 0; t < T_STEPS; t++) {
        // stage x_t tile: lane -> (node i = lane/2, dim d = lane%2)
        {
            int i = lane >> 1, d = lane & 1;
            int node = base + i;
            float v = (node < N) ? history[(size_t)node * (T_STEPS * 2) + t * 2 + d] : 0.0f;
            sX[wave][i][d] = v;
        }
        __syncthreads();

        // A fragments from h: four contiguous 8-half runs -> 2x v16h
        const _Float16* hrow = &sH[wave][nsel][0];
        v8h p0 = *(const v8h*)(hrow + kb);
        v8h p1 = *(const v8h*)(hrow + kb + 16);
        v8h p2 = *(const v8h*)(hrow + kb + 32);
        v8h p3 = *(const v8h*)(hrow + kb + 48);
        v16h a0 = CAT16(p0, p1);
        v16h a1 = CAT16(p2, p3);

        // 16 column tiles of the 256-wide gate matrix
        for (int ct = 0; ct < 16; ct++) {
            int col = ct * 16 + nsel;
            v8f acc;
#pragma unroll
            for (int r = 0; r < 8; r++) {
                int row = r + rowoff;
                acc[r] = sB[col] + sWih[col * 2 + 0] * sX[wave][row][0]
                                 + sWih[col * 2 + 1] * sX[wave][row][1];
            }
            v16h b0 = *(const v16h*)&sBfrag[ct][0][lane][0];
            v16h b1 = *(const v16h*)&sBfrag[ct][1][lane][0];
            acc = wmma_f16(a0, b0, acc);
            acc = wmma_f16(a1, b1, acc);
#pragma unroll
            for (int r = 0; r < 8; r++)
                sG[wave][r + rowoff][col] = (_Float16)acc[r];
        }
        __syncthreads();

        // gate activations + state update (torch gate order i,f,g,o)
        {
            const _Float16* gp = &sG[wave][nsel][0];
            v8h vi[4], vf[4], vgt[4], vo[4];
#pragma unroll
            for (int q = 0; q < 4; q++) {
                vi[q]  = *(const v8h*)(gp +       cc0 + q * 8);
                vf[q]  = *(const v8h*)(gp +  64 + cc0 + q * 8);
                vgt[q] = *(const v8h*)(gp + 128 + cc0 + q * 8);
                vo[q]  = *(const v8h*)(gp + 192 + cc0 + q * 8);
            }
            v8h hout[4];
            float hf[32];
#pragma unroll
            for (int j = 0; j < 32; j++) {
                int q = j >> 3, r = j & 7;
                float i_ = sigmoidf_((float)vi[q][r]);
                float f_ = sigmoidf_((float)vf[q][r]);
                float g_ = tanhf((float)vgt[q][r]);
                float o_ = sigmoidf_((float)vo[q][r]);
                creg[j] = f_ * creg[j] + i_ * g_;
                float h = o_ * tanhf(creg[j]);
                hout[q][r] = (_Float16)h;
                hf[j] = h;
            }
#pragma unroll
            for (int q = 0; q < 4; q++)
                *(v8h*)(&sH[wave][nsel][cc0 + q * 8]) = hout[q];
            if (t == T_STEPS - 1) {
                int node = base + nsel;
                if (node < N) {
#pragma unroll
                    for (int q = 0; q < 8; q++)
                        *(float4*)(&x_out[(size_t)node * H + cc0 + q * 4]) =
                            make_float4(hf[q * 4], hf[q * 4 + 1], hf[q * 4 + 2], hf[q * 4 + 3]);
                }
            }
        }
        __syncthreads();
    }
}

// ---------------------------------------------------------------------------
// Kernel 2: GATv2 node transform  xl = x@Wl+bl, xr = x@Wr+br  (WMMA)
// 4 waves/block, 16 nodes per wave.  Weights fragment-major in LDS.
// ---------------------------------------------------------------------------
__global__ __launch_bounds__(128)
void gat_transform(const float* __restrict__ x,
                   const float* __restrict__ Wl, const float* __restrict__ bl,
                   const float* __restrict__ Wr, const float* __restrict__ br,
                   float* __restrict__ xl, float* __restrict__ xr, int N)
{
    __shared__ _Float16 sBfrag[2][4][2][32][16];     // 16KB [m][ct][chunk][lane][j]
    __shared__ float    sBias[2][H];

    for (int idx = threadIdx.x; idx < 2 * 4 * 2 * 32 * 16; idx += 128) {
        int j     = idx & 15;
        int lane_ = (idx >> 4) & 31;
        int chunk = (idx >> 9) & 1;
        int ct    = (idx >> 10) & 3;
        int m     = idx >> 12;
        int col   = ct * 16 + (lane_ & 15);
        int k     = chunk * 32 + ((lane_ < 16) ? 0 : 8) + (j & 7) + (j >> 3) * 16;
        const float* W = m ? Wr : Wl;
        sBfrag[m][ct][chunk][lane_][j] = (_Float16)W[k * H + col];  // B[k][n]=W[k*64+n]
    }
    if (threadIdx.x < H) {
        sBias[0][threadIdx.x] = bl[threadIdx.x];
        sBias[1][threadIdx.x] = br[threadIdx.x];
    }
    __syncthreads();

    const int wave = threadIdx.x >> 5, lane = threadIdx.x & 31;
    const int base = (blockIdx.x * 4 + wave) * 16;
    const int nsel   = lane & 15;
    const int rowoff = (lane < 16) ? 0 : 8;
    const int kb     = (lane < 16) ? 0 : 8;

    // A fragments from global x (L2-resident), float4 runs -> f16
    v16h a0, a1;
    {
        const int anode = base + nsel;
        float buf[32];
        if (anode < N) {
            const float* xp = x + (size_t)anode * H;
#pragma unroll
            for (int rr = 0; rr < 4; rr++) {
                float4 lo = *(const float4*)(xp + kb + rr * 16);
                float4 hi = *(const float4*)(xp + kb + rr * 16 + 4);
                buf[rr * 8 + 0] = lo.x; buf[rr * 8 + 1] = lo.y;
                buf[rr * 8 + 2] = lo.z; buf[rr * 8 + 3] = lo.w;
                buf[rr * 8 + 4] = hi.x; buf[rr * 8 + 5] = hi.y;
                buf[rr * 8 + 6] = hi.z; buf[rr * 8 + 7] = hi.w;
            }
        } else {
#pragma unroll
            for (int j = 0; j < 32; j++) buf[j] = 0.0f;
        }
#pragma unroll
        for (int j = 0; j < 16; j++) {
            a0[j] = (_Float16)buf[j];        // runs 0,1 -> K = kb+{0..7,16..23}
            a1[j] = (_Float16)buf[16 + j];   // runs 2,3 -> +32
        }
    }

    for (int m = 0; m < 2; m++) {
        float* dst = m ? xr : xl;
        for (int ct = 0; ct < 4; ct++) {
            int col = ct * 16 + nsel;
            v8f acc;
#pragma unroll
            for (int r = 0; r < 8; r++) acc[r] = sBias[m][col];
            v16h b0 = *(const v16h*)&sBfrag[m][ct][0][lane][0];
            v16h b1 = *(const v16h*)&sBfrag[m][ct][1][lane][0];
            acc = wmma_f16(a0, b0, acc);
            acc = wmma_f16(a1, b1, acc);
#pragma unroll
            for (int r = 0; r < 8; r++) {
                int node = base + r + rowoff;
                if (node < N) dst[(size_t)node * H + col] = acc[r];
            }
        }
    }
}

// ---------------------------------------------------------------------------
// Edge-phase kernels
// ---------------------------------------------------------------------------
__global__ void gat_init(unsigned* __restrict__ mKey, float* __restrict__ denom,
                         float* __restrict__ accum, int N)
{
    int i = blockIdx.x * blockDim.x + threadIdx.x;
    if (i < N * H) accum[i] = 0.0f;
    if (i < N * 4) { mKey[i] = 0u; denom[i] = 0.0f; }
}

// pass 1: score per edge/head + segment max (ordered-uint atomicMax)
__global__ void gat_scores(const float* __restrict__ xl, const float* __restrict__ xr,
                           const int* __restrict__ ei, const float* __restrict__ att,
                           float* __restrict__ scores, unsigned* __restrict__ mKey,
                           int E, int EE)
{
    __shared__ float sAtt[H];
    if (threadIdx.x < H) sAtt[threadIdx.x] = att[threadIdx.x];
    __syncthreads();
    int e = blockIdx.x * blockDim.x + threadIdx.x;
    if (e >= EE) return;
    int s, d;
    if (e < E) { s = ei[e]; d = ei[E + e]; } else { s = d = e - E; }
    const float4* pl = (const float4*)(xl + (size_t)s * H);
    const float4* pr = (const float4*)(xr + (size_t)d * H);
    float sc[4] = {0.f, 0.f, 0.f, 0.f};
#pragma unroll
    for (int c4 = 0; c4 < 16; c4++) {
        float4 a = pl[c4];
        float4 b = pr[c4];
        int h = c4 >> 2;
        float v0 = a.x + b.x; v0 = (v0 > 0.f) ? v0 : 0.2f * v0;
        float v1 = a.y + b.y; v1 = (v1 > 0.f) ? v1 : 0.2f * v1;
        float v2 = a.z + b.z; v2 = (v2 > 0.f) ? v2 : 0.2f * v2;
        float v3 = a.w + b.w; v3 = (v3 > 0.f) ? v3 : 0.2f * v3;
        sc[h] += v0 * sAtt[c4 * 4 + 0] + v1 * sAtt[c4 * 4 + 1]
               + v2 * sAtt[c4 * 4 + 2] + v3 * sAtt[c4 * 4 + 3];
    }
#pragma unroll
    for (int h = 0; h < 4; h++) {
        scores[(size_t)e * 4 + h] = sc[h];
        atomicMax(&mKey[d * 4 + h], f2ord(sc[h]));
    }
}

// pass 2 (fused): alpha = exp(score - max[dst]); denom += alpha; accum += alpha*xl[src]
// (normalization by denom is deferred to finalize, so both sums can run together)
__global__ void gat_alpha_scatter(const float* __restrict__ scores, const int* __restrict__ ei,
                                  const unsigned* __restrict__ mKey,
                                  const float* __restrict__ xl,
                                  float* __restrict__ denom, float* __restrict__ accum,
                                  int E, int EE)
{
    int e = blockIdx.x * blockDim.x + threadIdx.x;
    if (e >= EE) return;
    int s, d;
    if (e < E) { s = ei[e]; d = ei[E + e]; } else { s = d = e - E; }
    float alpha[4];
#pragma unroll
    for (int h = 0; h < 4; h++) {
        float m = ord2f(mKey[d * 4 + h]);
        float a = __expf(scores[(size_t)e * 4 + h] - m);
        alpha[h] = a;
        atomicAdd(&denom[d * 4 + h], a);
    }
    const float4* pl = (const float4*)(xl + (size_t)s * H);
    float* pa = accum + (size_t)d * H;
#pragma unroll
    for (int c4 = 0; c4 < 16; c4++) {
        float4 v = pl[c4];
        float a = alpha[c4 >> 2];
        atomicAdd(&pa[c4 * 4 + 0], a * v.x);
        atomicAdd(&pa[c4 * 4 + 1], a * v.y);
        atomicAdd(&pa[c4 * 4 + 2], a * v.z);
        atomicAdd(&pa[c4 * 4 + 3], a * v.w);
    }
}

// pass 3: normalize, add bias, relu -> overwrite x ; optionally fuse global_add_pool
__global__ void gat_finalize(const float* __restrict__ accum, const float* __restrict__ denom,
                             const float* __restrict__ bias, float* __restrict__ x,
                             const int* __restrict__ batch, float* __restrict__ scene, int N)
{
    int i = blockIdx.x * blockDim.x + threadIdx.x;
    if (i >= N * H) return;
    int n = i >> 6, cc = i & 63, h = cc >> 4;
    float v = accum[i] / denom[n * 4 + h] + bias[cc];
    v = fmaxf(v, 0.0f);
    x[i] = v;
    if (scene != nullptr)
        atomicAdd(&scene[(size_t)batch[n] * H + cc], v);
}

// ---------------------------------------------------------------------------
// Misc + decoder
// ---------------------------------------------------------------------------
__global__ void zero_buf(float* __restrict__ p, int n)
{
    int i = blockIdx.x * blockDim.x + threadIdx.x;
    if (i < n) p[i] = 0.0f;
}

__global__ __launch_bounds__(128)
void decoder_kernel(const float* __restrict__ x, const float* __restrict__ scene,
                    const int* __restrict__ focal,
                    const float* __restrict__ W1, const float* __restrict__ b1,
                    const float* __restrict__ W2, const float* __restrict__ b2,
                    float* __restrict__ out)
{
    __shared__ float fused[128];
    __shared__ float hid[128];
    int g = blockIdx.x, t = threadIdx.x;
    int f = focal[g];
    fused[t] = (t < H) ? x[(size_t)f * H + t] : scene[(size_t)g * H + (t - H)];
    __syncthreads();
    float a = b1[t];
#pragma unroll 8
    for (int k = 0; k < 128; k++) a += fused[k] * W1[k * 128 + t];
    hid[t] = fmaxf(a, 0.0f);
    __syncthreads();
    if (t < HORIZON) {
        float o = b2[t];
#pragma unroll 8
        for (int k = 0; k < 128; k++) o += hid[k] * W2[k * HORIZON + t];
        out[(size_t)g * HORIZON + t] = o;
    }
}

// ---------------------------------------------------------------------------
// Launcher
// ---------------------------------------------------------------------------
extern "C" void kernel_launch(void* const* d_in, const int* in_sizes, int n_in,
                              void* d_out, int out_size, void* d_ws, size_t ws_size,
                              hipStream_t stream)
{
    (void)n_in; (void)out_size; (void)ws_size;
    const float* history    = (const float*)d_in[0];
    const int*   edge_index = (const int*)d_in[1];
    const int*   batch_idx  = (const int*)d_in[2];
    const int*   focal      = (const int*)d_in[3];
    const float* W_ih = (const float*)d_in[4];
    const float* W_hh = (const float*)d_in[5];
    const float* b_ih = (const float*)d_in[6];
    const float* b_hh = (const float*)d_in[7];
    const float* dec_W1 = (const float*)d_in[20];
    const float* dec_b1 = (const float*)d_in[21];
    const float* dec_W2 = (const float*)d_in[22];
    const float* dec_b2 = (const float*)d_in[23];

    const int N  = in_sizes[0] / (T_STEPS * 2);
    const int E  = in_sizes[1] / 2;
    const int G  = in_sizes[3];
    const int EE = E + N;

    size_t off = 0;
    auto wsa = [&](size_t bytes) -> char* {
        char* p = (char*)d_ws + off;
        off += (bytes + 255) & ~(size_t)255;
        return p;
    };
    float*    x      = (float*)wsa((size_t)N * H * sizeof(float));
    float*    xl     = (float*)wsa((size_t)N * H * sizeof(float));
    float*    xr     = (float*)wsa((size_t)N * H * sizeof(float));
    float*    accum  = (float*)wsa((size_t)N * H * sizeof(float));
    unsigned* mKey   = (unsigned*)wsa((size_t)N * 4 * sizeof(unsigned));
    float*    denom  = (float*)wsa((size_t)N * 4 * sizeof(float));
    float*    scores = (float*)wsa((size_t)EE * 4 * sizeof(float));
    float*    scene  = (float*)wsa((size_t)G * H * sizeof(float));

    // 1) LSTM encode (+ zero the pooled-scene buffer early, off critical path)
    lstm_kernel<<<(N + LSTM_WAVES * 16 - 1) / (LSTM_WAVES * 16), LSTM_WAVES * 32, 0, stream>>>(
        history, W_ih, W_hh, b_ih, b_hh, x, N);
    zero_buf<<<(G * H + 255) / 256, 256, 0, stream>>>(scene, G * H);

    // 2) two GATv2 layers (pool fused into the second finalize)
    const int nodeGrid = (N * H + 255) / 256;
    const int edgeGrid = (EE + 255) / 256;
    for (int layer = 0; layer < 2; layer++) {
        const float* Wl   = (const float*)d_in[8 + 6 * layer + 0];
        const float* bl   = (const float*)d_in[8 + 6 * layer + 1];
        const float* Wr   = (const float*)d_in[8 + 6 * layer + 2];
        const float* br   = (const float*)d_in[8 + 6 * layer + 3];
        const float* att  = (const float*)d_in[8 + 6 * layer + 4];
        const float* bias = (const float*)d_in[8 + 6 * layer + 5];

        gat_init<<<nodeGrid, 256, 0, stream>>>(mKey, denom, accum, N);
        gat_transform<<<(N + 63) / 64, 128, 0, stream>>>(x, Wl, bl, Wr, br, xl, xr, N);
        gat_scores<<<edgeGrid, 256, 0, stream>>>(xl, xr, edge_index, att, scores, mKey, E, EE);
        gat_alpha_scatter<<<edgeGrid, 256, 0, stream>>>(scores, edge_index, mKey, xl,
                                                        denom, accum, E, EE);
        gat_finalize<<<nodeGrid, 256, 0, stream>>>(accum, denom, bias, x,
                                                   batch_idx, (layer == 1) ? scene : nullptr, N);
    }

    // 3) decode
    decoder_kernel<<<G, 128, 0, stream>>>(x, scene, focal, dec_W1, dec_b1, dec_W2, dec_b2,
                                          (float*)d_out);
}
